// LoRAAttention_3435973837105
// MI455X (gfx1250) — matrix operand order repaired and checked by reading, not compiled, not run on hardware
//
#include <hip/hip_runtime.h>
#include <hip/hip_bf16.h>
#include <math.h>

// Problem constants (match reference)
constexpr int NB  = 2;     // batch
constexpr int SEQ = 4096;  // sequence
constexpr int DIM = 768;   // model dim
constexpr int RNK = 32;    // LoRA rank

typedef __attribute__((ext_vector_type(16))) __bf16 bf16x16;
typedef __attribute__((ext_vector_type(8)))  float  f32x8;

__device__ __forceinline__ unsigned short f2bf(float f) {
    unsigned int x = __float_as_uint(f);
    unsigned int r = (x + 0x7FFFu + ((x >> 16) & 1u)) >> 16;
    return (unsigned short)r;
}

// Async DMA: global (16B) -> LDS, tracked by ASYNCcnt.
// Generic LDS pointer low 32 bits == workgroup-relative LDS byte address
// (ISA 10.2 aperture rules), which is what VDST of the async op expects.
__device__ __forceinline__ void async_copy16(void* lds, const void* gptr) {
    unsigned ldsoff = (unsigned)(size_t)lds;
    unsigned long long ga = (unsigned long long)(size_t)gptr;
    asm volatile("global_load_async_to_lds_b128 %0, %1, off"
                 :: "v"(ldsoff), "v"(ga)
                 : "memory");
}
__device__ __forceinline__ void wait_async0() {
    asm volatile("s_wait_asynccnt 0" ::: "memory");
}

// ---------------------------------------------------------------- cvt f32->bf16
__global__ void cvt_f32_bf16(const float* __restrict__ in,
                             unsigned short* __restrict__ out, int n) {
    int i = blockIdx.x * 256 + threadIdx.x;
    if (i < n) out[i] = f2bf(in[i]);
}

// ------------------------------------------------------- fold LoRA into weights
// W_eff[n][k] = W[n][k] + sum_r Bm[n][r] * A[r][k]   (ALPHA = 1)
__global__ void merge_weights(const float* __restrict__ W,
                              const float* __restrict__ A,
                              const float* __restrict__ Bm,
                              unsigned short* __restrict__ out) {
    int i = blockIdx.x * 256 + threadIdx.x;          // i < DIM*DIM
    int n = i / DIM, k = i % DIM;
    float acc = W[i];
#pragma unroll
    for (int r = 0; r < RNK; ++r)
        acc += Bm[n * RNK + r] * A[r * DIM + k];
    out[i] = f2bf(acc);
}

// ------------------------------------------------------------- projection GEMM
// out[m][n] = sum_k xb[m][k] * wb[n][k] + bias[n]
// z=0 -> Q (row major), z=1 -> K (row major), z=2 -> V stored transposed [b][d][s]
__global__ void __launch_bounds__(256, 2)
proj_gemm(const unsigned short* __restrict__ xb,
          const unsigned short* __restrict__ wball,
          const float* __restrict__ bq, const float* __restrict__ bk,
          const float* __restrict__ bv,
          unsigned short* __restrict__ qb, unsigned short* __restrict__ kbuf,
          unsigned short* __restrict__ vT) {
    const int z = blockIdx.z;
    const unsigned short* wb = wball + (size_t)z * DIM * DIM;
    const float* bias = (z == 0) ? bq : (z == 1) ? bk : bv;
    unsigned short* outp = (z == 0) ? qb : (z == 1) ? kbuf : vT;

    const int m0 = blockIdx.y * 128;
    const int n0 = blockIdx.x * 64;

    __shared__ __align__(16) unsigned short As[128 * 32];
    __shared__ __align__(16) unsigned short Bs[64 * 32];

    const int tid  = threadIdx.x;
    const int lane = tid & 31;
    const int wave = tid >> 5;
    const int wm = wave >> 1;        // 0..3 -> 32-row slice
    const int wn = wave & 1;         // 0..1 -> 32-col slice
    const int half = lane >> 4;
    const int l16  = lane & 15;

    f32x8 acc[2][2] = {};

    for (int k0 = 0; k0 < DIM; k0 += 32) {
        // stage A: 128x32 bf16, async DMA straight into LDS (512 x 16B)
#pragma unroll
        for (int i = 0; i < 2; ++i) {
            int cid = tid * 2 + i;
            int row = cid >> 2, q = cid & 3;
            async_copy16(&As[row * 32 + q * 8],
                         &xb[(size_t)(m0 + row) * DIM + k0 + q * 8]);
        }
        // stage B: 64x32 bf16 (256 x 16B)
        {
            int row = tid >> 2, q = tid & 3;
            async_copy16(&Bs[row * 32 + q * 8],
                         &wb[(size_t)(n0 + row) * DIM + k0 + q * 8]);
        }
        wait_async0();
        __syncthreads();

#pragma unroll
        for (int mf = 0; mf < 2; ++mf) {
            int arow = wm * 32 + mf * 16 + l16;
            bf16x16 a;
            ((uint4*)&a)[0] = *(const uint4*)&As[arow * 32 + half * 8];
            ((uint4*)&a)[1] = *(const uint4*)&As[arow * 32 + 16 + half * 8];
#pragma unroll
            for (int nf = 0; nf < 2; ++nf) {
                int brow = wn * 32 + nf * 16 + l16;
                bf16x16 bfr;
                ((uint4*)&bfr)[0] = *(const uint4*)&Bs[brow * 32 + half * 8];
                ((uint4*)&bfr)[1] = *(const uint4*)&Bs[brow * 32 + 16 + half * 8];
                acc[mf][nf] = __builtin_amdgcn_wmma_f32_16x16x32_bf16(
                    false, a, false, bfr, (short)0, acc[mf][nf], false, false);
            }
        }
        __syncthreads();
    }

    // epilogue: bias + bf16 store (V transposed)
#pragma unroll
    for (int mf = 0; mf < 2; ++mf)
#pragma unroll
        for (int nf = 0; nf < 2; ++nf)
#pragma unroll
            for (int v = 0; v < 8; ++v) {
                int m = m0 + wm * 32 + mf * 16 + v + half * 8;
                int n = n0 + wn * 32 + nf * 16 + l16;
                float val = acc[mf][nf][v] + bias[n];
                unsigned short h = f2bf(val);
                if (z < 2) {
                    outp[(size_t)m * DIM + n] = h;
                } else {
                    int bidx = m >> 12;       // / SEQ
                    int s    = m & 4095;      // % SEQ
                    outp[(size_t)bidx * DIM * SEQ + (size_t)n * SEQ + s] = h;
                }
            }
}

// ------------------------------------------------------------- flash attention
// One block: batch b, 32 query rows. 8 waves. Streams 64-key tiles.
__global__ void __launch_bounds__(256, 1)
flash_attn(const unsigned short* __restrict__ qb,
           const unsigned short* __restrict__ kbuf,
           const unsigned short* __restrict__ vT,
           float* __restrict__ out) {
    const int b  = blockIdx.y;
    const int q0 = blockIdx.x * 32;
    const float scale = 0.03608439182435161f;   // 1/sqrt(768)

    __shared__ __align__(16) unsigned short Qs[32 * DIM];   // 48 KB
    __shared__ __align__(16) float          Ss[32 * 64];    //  8 KB
    __shared__ __align__(16) unsigned short Ps[32 * 64];    //  4 KB
    __shared__ float rowscale[32];
    __shared__ float rowlinv[32];

    const int tid  = threadIdx.x;
    const int lane = tid & 31;
    const int wave = tid >> 5;
    const int half = lane >> 4;
    const int l16  = lane & 15;

    const unsigned short* qbase = qb   + (size_t)b * SEQ * DIM;
    const unsigned short* kbase = kbuf + (size_t)b * SEQ * DIM;
    const unsigned short* vbase = vT   + (size_t)b * DIM * SEQ;

    // load Q tile (32 x 768 bf16) via async DMA: 3072 x 16B / 256 threads
#pragma unroll
    for (int i = 0; i < 12; ++i) {
        int cid = tid + i * 256;
        int row = cid / 96;      // 96 chunks per row
        int q   = cid % 96;
        async_copy16(&Qs[row * DIM + q * 8],
                     &qbase[(size_t)(q0 + row) * DIM + q * 8]);
    }
    wait_async0();
    __syncthreads();

    float m_run = -3.0e38f, l_run = 0.0f;   // used by threads 0..31 (wave 0)

    f32x8 oacc[2][6] = {};
    const int sm = wave & 1;       // score frag row  (0..1) -> 32 rows
    const int sn = wave >> 1;      // score frag col  (0..3) -> 64 keys
    const int ncol0 = wave * 96;   // this wave's 96-wide slice of D for O

    for (int key0 = 0; key0 < SEQ; key0 += 64) {
        // ---- prefetch next tile into L2/L0 ----
        if (key0 + 64 < SEQ) {
            __builtin_prefetch(&kbase[(size_t)(key0 + 64 + sn * 16 + l16) * DIM], 0, 1);
            __builtin_prefetch(&vbase[(size_t)(ncol0 + l16) * SEQ + key0 + 64], 0, 1);
        }

        // ---- scores: S = (Q K^T) * scale, each wave one 16x16 frag ----
        f32x8 sacc = {};
        const unsigned short* krow =
            kbase + (size_t)(key0 + sn * 16 + l16) * DIM;
        for (int kd = 0; kd < DIM; kd += 32) {
            bf16x16 a, bb;
            int arow = sm * 16 + l16;
            ((uint4*)&a)[0] = *(const uint4*)&Qs[arow * DIM + kd + half * 8];
            ((uint4*)&a)[1] = *(const uint4*)&Qs[arow * DIM + kd + 16 + half * 8];
            const uint4* kp = (const uint4*)(krow + kd);
            ((uint4*)&bb)[0] = kp[half];
            ((uint4*)&bb)[1] = kp[2 + half];
            sacc = __builtin_amdgcn_wmma_f32_16x16x32_bf16(
                false, a, false, bb, (short)0, sacc, false, false);
        }
#pragma unroll
        for (int v = 0; v < 8; ++v)
            Ss[(sm * 16 + v + half * 8) * 64 + sn * 16 + l16] = sacc[v] * scale;
        __syncthreads();

        // ---- online softmax: one thread per row (wave 0) ----
        if (tid < 32) {
            int row = tid;
            float mt = -3.0e38f;
            for (int j = 0; j < 64; ++j) mt = fmaxf(mt, Ss[row * 64 + j]);
            float mnew = fmaxf(m_run, mt);
            float corr = __expf(m_run - mnew);
            float s = 0.0f;
            for (int j = 0; j < 64; ++j) {
                float p = __expf(Ss[row * 64 + j] - mnew);
                s += p;
                Ps[row * 64 + j] = f2bf(p);
            }
            l_run = l_run * corr + s;
            m_run = mnew;
            rowscale[row] = corr;
        }
        __syncthreads();

        // ---- rescale O accumulators ----
#pragma unroll
        for (int mf = 0; mf < 2; ++mf)
#pragma unroll
            for (int v = 0; v < 8; ++v) {
                float c = rowscale[mf * 16 + v + half * 8];
#pragma unroll
                for (int nf = 0; nf < 6; ++nf) oacc[mf][nf][v] *= c;
            }

        // ---- O += P @ V : per wave, 32 rows x 96 cols over 64 keys ----
#pragma unroll
        for (int ks = 0; ks < 2; ++ks) {
            bf16x16 ap[2];
#pragma unroll
            for (int mf = 0; mf < 2; ++mf) {
                int prow = mf * 16 + l16;
                ((uint4*)&ap[mf])[0] =
                    *(const uint4*)&Ps[prow * 64 + ks * 32 + half * 8];
                ((uint4*)&ap[mf])[1] =
                    *(const uint4*)&Ps[prow * 64 + ks * 32 + 16 + half * 8];
            }
#pragma unroll
            for (int nf = 0; nf < 6; ++nf) {
                int dcol = ncol0 + nf * 16 + l16;
                const uint4* vp = (const uint4*)
                    &vbase[(size_t)dcol * SEQ + key0 + ks * 32];
                bf16x16 bv;
                ((uint4*)&bv)[0] = vp[half];
                ((uint4*)&bv)[1] = vp[2 + half];
#pragma unroll
                for (int mf = 0; mf < 2; ++mf)
                    oacc[mf][nf] = __builtin_amdgcn_wmma_f32_16x16x32_bf16(
                        false, ap[mf], false, bv, (short)0, oacc[mf][nf],
                        false, false);
            }
        }
        __syncthreads();
    }

    if (tid < 32) rowlinv[tid] = 1.0f / l_run;
    __syncthreads();

    float* obase = out + (size_t)b * SEQ * DIM + (size_t)q0 * DIM;
#pragma unroll
    for (int mf = 0; mf < 2; ++mf)
#pragma unroll
        for (int v = 0; v < 8; ++v) {
            int row = mf * 16 + v + half * 8;
            float inv = rowlinv[row];
#pragma unroll
            for (int nf = 0; nf < 6; ++nf) {
                int dcol = ncol0 + nf * 16 + l16;
                obase[(size_t)row * DIM + dcol] = oacc[mf][nf][v] * inv;
            }
        }
}

// -------------------------------------------------------------------- launcher
extern "C" void kernel_launch(void* const* d_in, const int* in_sizes, int n_in,
                              void* d_out, int out_size, void* d_ws,
                              size_t ws_size, hipStream_t stream) {
    const float* x  = (const float*)d_in[0];
    const float* Wq = (const float*)d_in[1];
    const float* bq = (const float*)d_in[2];
    const float* Wk = (const float*)d_in[3];
    const float* bk = (const float*)d_in[4];
    const float* Wv = (const float*)d_in[5];
    const float* bv = (const float*)d_in[6];
    const float* Aq = (const float*)d_in[7];
    const float* Bq = (const float*)d_in[8];
    const float* Ak = (const float*)d_in[9];
    const float* Bk = (const float*)d_in[10];
    const float* Av = (const float*)d_in[11];
    const float* Bv = (const float*)d_in[12];
    float* out = (float*)d_out;

    char* ws = (char*)d_ws;
    size_t off = 0;
    auto alloc = [&](size_t bytes) {
        void* p = ws + off;
        off += (bytes + 255) & ~(size_t)255;
        return p;
    };
    const size_t BSD = (size_t)NB * SEQ * DIM;
    unsigned short* xb = (unsigned short*)alloc(BSD * 2);
    unsigned short* wb = (unsigned short*)alloc((size_t)3 * DIM * DIM * 2);
    unsigned short* qb = (unsigned short*)alloc(BSD * 2);
    unsigned short* kb = (unsigned short*)alloc(BSD * 2);
    unsigned short* vT = (unsigned short*)alloc(BSD * 2);

    cvt_f32_bf16<<<(int)(BSD / 256), 256, 0, stream>>>(x, xb, (int)BSD);

    merge_weights<<<(DIM * DIM) / 256, 256, 0, stream>>>(Wq, Aq, Bq, wb);
    merge_weights<<<(DIM * DIM) / 256, 256, 0, stream>>>(Wk, Ak, Bk,
                                                         wb + DIM * DIM);
    merge_weights<<<(DIM * DIM) / 256, 256, 0, stream>>>(Wv, Av, Bv,
                                                         wb + 2 * DIM * DIM);

    dim3 gp(DIM / 64, (NB * SEQ) / 128, 3);
    proj_gemm<<<gp, 256, 0, stream>>>(xb, wb, bq, bk, bv, qb, kb, vT);

    dim3 ga(SEQ / 32, NB);
    flash_attn<<<ga, 256, 0, stream>>>(qb, kb, vT, out);
}
